// MultiHeadAttention_75935021794134
// MI455X (gfx1250) — compile-verified, matrix operand order checked by reading, not run on hardware
//
#include <hip/hip_runtime.h>
#include <hip/hip_bf16.h>

// ---------------------------------------------------------------------------
// MI455X (gfx1250) MHA forward.
//   - all matmuls via v_wmma_f32_16x16x32_f16 (wave32, fp32 accumulate)
//   - flash-attention with online softmax (no S x S score materialization)
//   - K/V tiles staged with GLOBAL_LOAD_ASYNC_TO_LDS_B128 (ASYNCcnt-tracked,
//     double-buffered: tile t+1 copies overlap tile t compute)
//   - mask (all-zero in reference, but read for fidelity) hits the 192MB L2
// Fragment layouts per CDNA5 ISA 7.12.2:
//   A (16x32 f16): lane L -> row L&15, K(e) = (e&7) + 16*(e>>3) + 8*(L>>4)
//   B (32x16 f16): lane L -> col L&15, same K mapping
//   C/D (16x16 f32, 8 VGPR): elem j -> row j + 8*(L>>4), col L&15
// Consecutive element pairs are K-contiguous -> fragments load as 8x b32
// (compiler merges them into ds_load_2addr_b64).
// ---------------------------------------------------------------------------

typedef __attribute__((ext_vector_type(16))) _Float16 v16h;
typedef __attribute__((ext_vector_type(8)))  float    v8f;

__device__ __forceinline__ v8f wmma_f16(v16h a, v16h b, v8f c) {
  return __builtin_amdgcn_wmma_f32_16x16x32_f16(false, a, false, b, (short)0, c,
                                                false, false);
}

// Gather one 16-half fragment as 8 packed 32-bit loads (pairs contiguous in K).
__device__ __forceinline__ v16h ld_frag(const _Float16* base, const int* koff) {
  union { unsigned u[8]; v16h h; } t;
#pragma unroll
  for (int p = 0; p < 8; ++p)
    t.u[p] = *(const unsigned*)(base + koff[p]);
  return t.h;
}

__device__ __forceinline__ v8f zero8() {
  v8f z = {0.f, 0.f, 0.f, 0.f, 0.f, 0.f, 0.f, 0.f};
  return z;
}

// Wave-relative LDS byte offset: flat LDS addresses carry the wave-relative
// offset in their low 32 bits (CDNA5 ISA 10.2 aperture mapping).
__device__ __forceinline__ unsigned lds_off(const void* p) {
  return (unsigned)(size_t)p;
}

// Async DMA: 16 bytes global -> LDS, tracked with ASYNCcnt (no VGPR data path).
__device__ __forceinline__ void async_copy_b128(unsigned lds_byte_off,
                                                const void* gaddr) {
  asm volatile("global_load_async_to_lds_b128 %0, %1, off"
               :
               : "v"(lds_byte_off), "v"(gaddr)
               : "memory");
}

__device__ __forceinline__ void wait_async0() {
  asm volatile("s_wait_asynccnt 0x0" ::: "memory");
}

// ---------------------------------------------------------------------------
// GEMM: C[4096x512] = X[4096x512] * W[512x512], f16 compute / f32 accumulate.
// Block tile 128x64, 8 waves (each wave: 16 rows x 64 cols = 4 accumulators).
// MODE 0: out f16 [H][S][64]      (Q, K per-head layout)
// MODE 1: out f16 [H][64][S]      (V transposed per-head layout)
// MODE 2: out f32 [S][512] + bias (final projection)
// ---------------------------------------------------------------------------
template <int MODE>
__global__ __launch_bounds__(256) void gemm512(const float* __restrict__ X,
                                               const float* __restrict__ W,
                                               const float* __restrict__ bias,
                                               _Float16* __restrict__ out16,
                                               float* __restrict__ out32) {
  constexpr int D = 512;
  const int m0 = blockIdx.x * 128;
  const int n0 = blockIdx.y * 64;
  const int t    = threadIdx.x;
  const int lane = t & 31;
  const int w    = t >> 5;      // wave id 0..7 (wave32)
  const int r    = lane & 15;
  const int g    = lane >> 4;

  __shared__ _Float16 As[128 * 34];  // [row][k], stride 34 halves
  __shared__ _Float16 Bs[64 * 34];   // transposed: [n][k], stride 34

  int koff[8];
#pragma unroll
  for (int p = 0; p < 8; ++p) koff[p] = 2 * (p & 3) + 16 * (p >> 2) + 8 * g;

  v8f acc[4];
#pragma unroll
  for (int c = 0; c < 4; ++c) acc[c] = zero8();

  for (int kt = 0; kt < 16; ++kt) {
    const int k0 = kt * 32;
    __syncthreads();
    // A tile 128x32 f32 -> f16 LDS, 4x float4 per thread
#pragma unroll
    for (int i = 0; i < 4; ++i) {
      int idx = t + 256 * i;              // 0..1023 float4s
      int row = idx >> 3, c4 = idx & 7;
      float4 v = *(const float4*)(X + (size_t)(m0 + row) * D + k0 + 4 * c4);
      union { unsigned u; _Float16 h[2]; } pa, pb;
      pa.h[0] = (_Float16)v.x; pa.h[1] = (_Float16)v.y;
      pb.h[0] = (_Float16)v.z; pb.h[1] = (_Float16)v.w;
      *(unsigned*)(As + row * 34 + 4 * c4)     = pa.u;
      *(unsigned*)(As + row * 34 + 4 * c4 + 2) = pb.u;
    }
    // B tile 32x64 f32 -> f16 LDS transposed [n][k], 2x float4 per thread
#pragma unroll
    for (int i = 0; i < 2; ++i) {
      int idx = t + 256 * i;              // 0..511 float4s
      int kr = idx >> 4, c4 = idx & 15;
      float4 v = *(const float4*)(W + (size_t)(k0 + kr) * D + n0 + 4 * c4);
      Bs[(4 * c4 + 0) * 34 + kr] = (_Float16)v.x;
      Bs[(4 * c4 + 1) * 34 + kr] = (_Float16)v.y;
      Bs[(4 * c4 + 2) * 34 + kr] = (_Float16)v.z;
      Bs[(4 * c4 + 3) * 34 + kr] = (_Float16)v.w;
    }
    __syncthreads();
    v16h af = ld_frag(As + (16 * w + r) * 34, koff);
#pragma unroll
    for (int c = 0; c < 4; ++c) {
      v16h bf = ld_frag(Bs + (16 * c + r) * 34, koff);
      acc[c] = wmma_f16(af, bf, acc[c]);
    }
  }

#pragma unroll
  for (int c = 0; c < 4; ++c) {
    const int n = n0 + 16 * c + r;
#pragma unroll
    for (int j = 0; j < 8; ++j) {
      const int m = m0 + 16 * w + j + 8 * g;
      float val = acc[c][j];
      if (MODE == 2) {
        out32[(size_t)m * D + n] = val + bias[n];
      } else if (MODE == 0) {
        int h = n >> 6, dh = n & 63;
        out16[((size_t)h * 4096 + m) * 64 + dh] = (_Float16)val;
      } else {
        int h = n >> 6, dh = n & 63;
        out16[((size_t)h * 64 + dh) * 4096 + m] = (_Float16)val;
      }
    }
  }
}

// ---------------------------------------------------------------------------
// Flash attention. Grid (32 qblocks, 8 heads), 256 threads (8 waves).
// Block handles 128 query rows; 64-key tiles double-buffered in LDS via
// GLOBAL_LOAD_ASYNC_TO_LDS_B128. Each wave owns 16 query rows.
// ---------------------------------------------------------------------------
__global__ __launch_bounds__(256) void flash_attn(const _Float16* __restrict__ Qh,
                                                  const _Float16* __restrict__ Kh,
                                                  const _Float16* __restrict__ VhT,
                                                  const float* __restrict__ mask,
                                                  float* __restrict__ ctx) {
  constexpr int S = 4096, DH = 64, D = 512;
  constexpr int KVS = 72;            // K/V row stride in halves: 144B, 16B-aligned
  constexpr int KVT = 64 * KVS;      // one K or V tile buffer (halves)
  constexpr int PS  = 68;            // P row stride (halves)

  const int head = blockIdx.y;
  const int m0   = blockIdx.x * 128;
  const int t    = threadIdx.x;
  const int lane = t & 31;
  const int w    = t >> 5;
  const int r    = lane & 15;
  const int g    = lane >> 4;

  __shared__ __align__(16) _Float16 smem[4 * KVT + 8 * 16 * PS];
  _Float16* Kt0 = smem;              // two K buffers
  _Float16* Vt0 = smem + 2 * KVT;    // two V buffers
  _Float16* Pt  = smem + 4 * KVT;    // per-wave P tiles [16][64]

  int koff[8];
#pragma unroll
  for (int p = 0; p < 8; ++p) koff[p] = 2 * (p & 3) + 16 * (p >> 2) + 8 * g;

  // Q A-fragments: rows m0+16w+r, two 32-wide K chunks over Dh=64 (in regs)
  const int qrow = m0 + 16 * w + r;
  v16h qa[2];
#pragma unroll
  for (int kk = 0; kk < 2; ++kk) {
    union { unsigned u[8]; v16h h; } tmp;
#pragma unroll
    for (int p = 0; p < 8; ++p)
      tmp.u[p] = *(const unsigned*)(Qh + ((size_t)head * S + qrow) * DH +
                                    kk * 32 + koff[p]);
    qa[kk] = tmp.h;
  }

  v8f oacc[4];
#pragma unroll
  for (int c = 0; c < 4; ++c) oacc[c] = zero8();
  float mrow[8], lrow[8];
#pragma unroll
  for (int j = 0; j < 8; ++j) { mrow[j] = -1.0e30f; lrow[j] = 0.0f; }

  const float scale = 0.125f;  // 1/(sqrt(64)+1e-9) rounds to 0.125 in f32

  // Issue async copies for one 64-key tile: per thread 2x16B of K + 2x16B of V.
  auto issue_tile = [&](int t0, int buf) {
#pragma unroll
    for (int i = 0; i < 2; ++i) {
      int idx = t + 256 * i;            // 0..511 chunks of 8 halves
      int row = idx >> 3, c8 = idx & 7;
      async_copy_b128(
          lds_off(Kt0 + buf * KVT + row * KVS + 8 * c8),
          (const void*)(Kh + ((size_t)head * S + t0 + row) * DH + 8 * c8));
      async_copy_b128(
          lds_off(Vt0 + buf * KVT + row * KVS + 8 * c8),
          (const void*)(VhT + ((size_t)head * DH + row) * S + t0 + 8 * c8));
    }
  };

  issue_tile(0, 0);  // prologue

  for (int t0 = 0; t0 < S; t0 += 64) {
    const int cur = (t0 >> 6) & 1;
    wait_async0();      // this wave's copies for tile t0 have landed in LDS
    __syncthreads();    // all waves landed; all waves done reading buf cur^1
    if (t0 + 64 < S) issue_tile(t0 + 64, cur ^ 1);  // overlaps compute below

    const _Float16* Kc = Kt0 + cur * KVT;
    const _Float16* Vc = Vt0 + cur * KVT;

    // S = Q * K^T : 2 K-chunks x 4 key-column blocks
    v8f sacc[4];
#pragma unroll
    for (int c = 0; c < 4; ++c) sacc[c] = zero8();
#pragma unroll
    for (int kk = 0; kk < 2; ++kk) {
#pragma unroll
      for (int c = 0; c < 4; ++c) {
        v16h kb = ld_frag(Kc + (16 * c + r) * KVS + kk * 32, koff);
        sacc[c] = wmma_f16(qa[kk], kb, sacc[c]);
      }
    }

    // Online softmax. Row j+8g is spread over the 16 lanes sharing (lane>>4);
    // xor-shuffles over {1,2,4,8} reduce within that half.
#pragma unroll
    for (int j = 0; j < 8; ++j) {
      const int qg = m0 + 16 * w + j + 8 * g;
      float s[4], mx = -1.0e30f;
#pragma unroll
      for (int c = 0; c < 4; ++c) {
        float mval = mask[(size_t)qg * S + t0 + 16 * c + r];
        s[c] = sacc[c][j] * scale + mval * (-1.0e9f);
        mx = fmaxf(mx, s[c]);
      }
      mx = fmaxf(mx, __shfl_xor(mx, 1, 32));
      mx = fmaxf(mx, __shfl_xor(mx, 2, 32));
      mx = fmaxf(mx, __shfl_xor(mx, 4, 32));
      mx = fmaxf(mx, __shfl_xor(mx, 8, 32));
      float mnew = fmaxf(mrow[j], mx);
      float corr = expf(mrow[j] - mnew);
      float ps = 0.0f;
#pragma unroll
      for (int c = 0; c < 4; ++c) {
        float pv = expf(s[c] - mnew);
        ps += pv;
        Pt[(w * 16 + j + 8 * g) * PS + 16 * c + r] = (_Float16)pv;
      }
      ps += __shfl_xor(ps, 1, 32);
      ps += __shfl_xor(ps, 2, 32);
      ps += __shfl_xor(ps, 4, 32);
      ps += __shfl_xor(ps, 8, 32);
      lrow[j] = lrow[j] * corr + ps;
      mrow[j] = mnew;
#pragma unroll
      for (int c = 0; c < 4; ++c) oacc[c][j] *= corr;
    }
    __syncthreads();  // P tile visible (cross-lane) before A-frag gather

    // O += P * V : P as A-frags (K = 64 keys), V as B-frags from [dh][key]
    v16h pa[2];
#pragma unroll
    for (int kk = 0; kk < 2; ++kk)
      pa[kk] = ld_frag(Pt + (w * 16 + r) * PS + kk * 32, koff);
#pragma unroll
    for (int kk = 0; kk < 2; ++kk) {
#pragma unroll
      for (int c = 0; c < 4; ++c) {
        v16h vb = ld_frag(Vc + (16 * c + r) * KVS + kk * 32, koff);
        oacc[c] = wmma_f16(pa[kk], vb, oacc[c]);
      }
    }
  }

  // Normalize and write ctx (heads merged): ctx[s][head*64 + dh], f32.
#pragma unroll
  for (int j = 0; j < 8; ++j) {
    const float inv = 1.0f / lrow[j];
    const int qg = m0 + 16 * w + j + 8 * g;
#pragma unroll
    for (int c = 0; c < 4; ++c)
      ctx[(size_t)qg * D + head * 64 + 16 * c + r] = oacc[c][j] * inv;
  }
}

// ---------------------------------------------------------------------------
// Launch: 3 projections -> flash attention -> output projection. 20MB ws.
// ---------------------------------------------------------------------------
extern "C" void kernel_launch(void* const* d_in, const int* in_sizes, int n_in,
                              void* d_out, int out_size, void* d_ws, size_t ws_size,
                              hipStream_t stream) {
  (void)in_sizes; (void)n_in; (void)out_size; (void)ws_size;
  const float* q    = (const float*)d_in[0];
  const float* k    = (const float*)d_in[1];
  const float* v    = (const float*)d_in[2];
  const float* mask = (const float*)d_in[3];
  const float* wq   = (const float*)d_in[4];
  const float* wk   = (const float*)d_in[5];
  const float* wv   = (const float*)d_in[6];
  const float* wo   = (const float*)d_in[7];
  const float* bo   = (const float*)d_in[8];
  float* out = (float*)d_out;

  char* ws = (char*)d_ws;
  _Float16* Qh  = (_Float16*)(ws + 0);                        // 4 MB [H][S][64]
  _Float16* Kh  = (_Float16*)(ws + (size_t)4  * 1024 * 1024); // 4 MB [H][S][64]
  _Float16* VhT = (_Float16*)(ws + (size_t)8  * 1024 * 1024); // 4 MB [H][64][S]
  float*    ctx = (float*)   (ws + (size_t)12 * 1024 * 1024); // 8 MB [S][512]

  dim3 ggrid(32, 8);   // 4096/128 row blocks x 512/64 col blocks
  dim3 gblk(256);
  gemm512<0><<<ggrid, gblk, 0, stream>>>(q, wq, nullptr, Qh, nullptr);
  gemm512<0><<<ggrid, gblk, 0, stream>>>(k, wk, nullptr, Kh, nullptr);
  gemm512<1><<<ggrid, gblk, 0, stream>>>(v, wv, nullptr, VhT, nullptr);
  flash_attn<<<dim3(32, 8), gblk, 0, stream>>>(Qh, Kh, VhT, mask, ctx);
  gemm512<2><<<ggrid, gblk, 0, stream>>>(ctx, wo, bo, nullptr, out);
}
